// SparseDeepCCA_21148418965905
// MI455X (gfx1250) — compile-verified, matrix operand order checked by reading, not run on hardware
//
#include <hip/hip_runtime.h>
#include <hip/hip_bf16.h>
#include <math.h>

// ---------------------------------------------------------------------------
// SparseDeepCCA loss on MI455X (gfx1250, wave32, WMMA bf16 16x16x32)
//
// Pipeline:
//  1) proj_gemm   : Xh = (X * gate(mu_x)) @ Wx + bx    (bf16 WMMA, f32 accum)
//  2) colsum      : per-column sums of Xh / Yh (for mean subtraction)
//  3) gram_kernel : split-K partial Grams Xh^T Xh, Xh^T Yh, Yh^T Yh (bf16 WMMA)
//  4) fixup       : reduce split-K partials, subtract mean outer product,
//                   scale by 1/(N-1), add eps (Cxx+e) / 2*eps (Cyy+2e) on diag
//  5) cholesky256 : in-place Cholesky of Axx, Ayy (single block)
//  6) chol_solve  : A1=(Cyy+2e)^{-1} Cyx,  A2=(Cxx+e)^{-1} Cxy (per-column trsm)
//  7) finalize    : loss = -trace(A1@A2)/D + STG regularizers (erf)
//
// trace identity used: tr(R Cyx Kxx Cxy R) = tr(R^2 Cyx Kxx Cxy), R^2=(Cyy+2e)^-1
// ---------------------------------------------------------------------------

typedef __attribute__((ext_vector_type(16))) __bf16 v16bf;
typedef __attribute__((ext_vector_type(8)))  __bf16 v8bf;
typedef __attribute__((ext_vector_type(2)))  __bf16 v2bf;
typedef __attribute__((ext_vector_type(8)))  float  v8f;

#define NSAMP  65536
#define DIN    1024
#define DPROJ  256
#define EPSR   0.001f
#define NKCHUNK 8
#define KCHUNK (NSAMP / NKCHUNK)

// Native bf16 conversion: gfx1250 has hardware f32->bf16 converts, so let the
// compiler emit v_cvt_pk_bf16_f32-class ops instead of a VALU bit-twiddle chain.
__device__ __forceinline__ __bf16 f2bf(float f) { return (__bf16)f; }

// Convert a pair together so the backend can pack into one cvt_pk instruction.
__device__ __forceinline__ v2bf f2bf2(float a, float b) {
  v2bf r; r[0] = (__bf16)a; r[1] = (__bf16)b; return r;
}

__device__ __forceinline__ float gateval(float mu) {
  return fminf(fmaxf(mu + 0.5f, 0.0f), 1.0f);
}

// Load one 16x32 bf16 A/B fragment from LDS stored [row][k] (k contiguous, 32 wide).
// Lane layout (ISA 7.12.2): lanes 0-15 hold K 0..7 / 16..23, lanes 16-31 K 8..15 / 24..31.
__device__ __forceinline__ v16bf ldfrag(const __bf16* rowbase, int khalf) {
  const v8bf lo = *(const v8bf*)(rowbase + khalf * 8);
  const v8bf hi = *(const v8bf*)(rowbase + 16 + khalf * 8);
  v16bf r;
#pragma unroll
  for (int i = 0; i < 8; ++i) { r[i] = lo[i]; r[i + 8] = hi[i]; }
  return r;
}

// ---------------------------------------------------------------------------
// 1) Gated projection GEMM:  Out[N,256] = (Xin * gate(mu)) @ W + bias
//    grid = (NSAMP/128, DPROJ/128), block = 256 threads (8 waves)
// ---------------------------------------------------------------------------
__global__ __launch_bounds__(256) void proj_gemm(
    const float* __restrict__ Xin, const float* __restrict__ mu,
    const float* __restrict__ W,   const float* __restrict__ bias,
    float* __restrict__ Out, int K) {
  const int rowBase = blockIdx.x * 128;
  const int colBase = blockIdx.y * 128;
  const int tid  = threadIdx.x;
  const int lane = tid & 31;
  const int wave = tid >> 5;         // 0..7
  const int wr   = wave >> 1;        // 0..3 -> M offset wr*32
  const int wc   = wave & 1;         // 0..1 -> N offset wc*64
  const int khalf = lane >> 4;
  const int lrow  = lane & 15;

  __shared__ __bf16 As[128 * 32];    // [m][k]
  __shared__ __bf16 Bs[128 * 32];    // [n][k]  (W tile stored transposed)

  v8f acc[2][4];
#pragma unroll
  for (int a = 0; a < 2; ++a)
#pragma unroll
    for (int b = 0; b < 4; ++b) acc[a][b] = (v8f){};

  for (int k0 = 0; k0 < K; k0 += 32) {
    __syncthreads();
    // --- load + gate + convert X tile [128 x 32] ---
#pragma unroll
    for (int r = 0; r < 4; ++r) {
      int f4  = tid + r * 256;             // 1024 float4 total
      int row = f4 >> 3;                   // 8 float4 per row
      int c4  = (f4 & 7) * 4;
      const float* src = Xin + (size_t)(rowBase + row) * K + k0 + c4;
      if (k0 + 32 < K) __builtin_prefetch(src + 32, 0, 0);  // global_prefetch next K-tile
      float4 v = *(const float4*)src;
      float g0 = gateval(mu[k0 + c4 + 0]);
      float g1 = gateval(mu[k0 + c4 + 1]);
      float g2 = gateval(mu[k0 + c4 + 2]);
      float g3 = gateval(mu[k0 + c4 + 3]);
      *(v2bf*)&As[row * 32 + c4 + 0] = f2bf2(v.x * g0, v.y * g1);
      *(v2bf*)&As[row * 32 + c4 + 2] = f2bf2(v.z * g2, v.w * g3);
    }
    // --- load + convert W tile [32 x 128], store transposed [n][k] ---
#pragma unroll
    for (int r = 0; r < 4; ++r) {
      int f4   = tid + r * 256;            // 1024 float4 total
      int krow = f4 >> 5;                  // 32 float4 per k-row
      int c4   = (f4 & 31) * 4;
      float4 v = *(const float4*)(W + (size_t)(k0 + krow) * DPROJ + colBase + c4);
      Bs[(c4 + 0) * 32 + krow] = f2bf(v.x);
      Bs[(c4 + 1) * 32 + krow] = f2bf(v.y);
      Bs[(c4 + 2) * 32 + krow] = f2bf(v.z);
      Bs[(c4 + 3) * 32 + krow] = f2bf(v.w);
    }
    __syncthreads();

    v16bf afr[2], bfr[4];
#pragma unroll
    for (int mt = 0; mt < 2; ++mt)
      afr[mt] = ldfrag(&As[(wr * 32 + mt * 16 + lrow) * 32], khalf);
#pragma unroll
    for (int nt = 0; nt < 4; ++nt)
      bfr[nt] = ldfrag(&Bs[(wc * 64 + nt * 16 + lrow) * 32], khalf);
#pragma unroll
    for (int mt = 0; mt < 2; ++mt)
#pragma unroll
      for (int nt = 0; nt < 4; ++nt)
        acc[mt][nt] = __builtin_amdgcn_wmma_f32_16x16x32_bf16(
            false, afr[mt], false, bfr[nt], (short)0, acc[mt][nt], false, false);
  }

  // epilogue: + bias, store f32. C layout: VGPR r -> M=r+8*khalf, N=lane&15.
#pragma unroll
  for (int mt = 0; mt < 2; ++mt)
#pragma unroll
    for (int nt = 0; nt < 4; ++nt) {
      int col = colBase + wc * 64 + nt * 16 + lrow;
      float bv = bias[col];
#pragma unroll
      for (int r = 0; r < 8; ++r) {
        int row = rowBase + wr * 32 + mt * 16 + r + 8 * khalf;
        Out[(size_t)row * DPROJ + col] = acc[mt][nt][r] + bv;
      }
    }
}

// ---------------------------------------------------------------------------
// 2) column sums (for mean subtraction): grid = 256 blocks, one column each
// ---------------------------------------------------------------------------
__global__ __launch_bounds__(256) void colsum(const float* __restrict__ M,
                                              float* __restrict__ s, int nrows) {
  __shared__ float red[256];
  int col = blockIdx.x;
  float p = 0.f;
  for (int r = threadIdx.x; r < nrows; r += 256)
    p += M[(size_t)r * DPROJ + col];
  red[threadIdx.x] = p; __syncthreads();
  for (int o = 128; o > 0; o >>= 1) {
    if (threadIdx.x < o) red[threadIdx.x] += red[threadIdx.x + o];
    __syncthreads();
  }
  if (threadIdx.x == 0) s[col] = red[0];
}

// ---------------------------------------------------------------------------
// 3) split-K Gram: Gpart[z] += A[:,64ti:+64]^T B[:,64tj:+64] over K-chunk z
//    grid = (4, 4, NKCHUNK), block = 256 (8 waves, 2 output tiles each)
// ---------------------------------------------------------------------------
__global__ __launch_bounds__(256) void gram_kernel(
    const float* __restrict__ A, const float* __restrict__ B,
    float* __restrict__ Gpart) {
  const int ti = blockIdx.x, tj = blockIdx.y;
  const size_t kStart = (size_t)blockIdx.z * KCHUNK;
  const int tid  = threadIdx.x;
  const int lane = tid & 31;
  const int wave = tid >> 5;
  const int tr   = wave >> 1;          // 0..3 tile row
  const int tcB  = (wave & 1) * 2;     // tile cols {tcB, tcB+1}
  const int khalf = lane >> 4;
  const int lrow  = lane & 15;

  __shared__ __bf16 LA[64 * 32];       // [p][k]
  __shared__ __bf16 LB[64 * 32];       // [q][k]

  v8f acc[2] = {(v8f){}, (v8f){}};

  for (int k0 = 0; k0 < KCHUNK; k0 += 32) {
    __syncthreads();
#pragma unroll
    for (int r = 0; r < 2; ++r) {
      int f4   = tid + r * 256;        // 512 float4 per matrix
      int krow = f4 >> 4;              // 16 float4 per k-row
      int c4   = (f4 & 15) * 4;
      const float* pa = A + (kStart + k0 + krow) * DPROJ + ti * 64 + c4;
      const float* pb = B + (kStart + k0 + krow) * DPROJ + tj * 64 + c4;
      if (k0 + 32 < KCHUNK) { __builtin_prefetch(pa + 32 * DPROJ, 0, 0); }
      float4 va = *(const float4*)pa;
      float4 vb = *(const float4*)pb;
      LA[(c4 + 0) * 32 + krow] = f2bf(va.x);
      LA[(c4 + 1) * 32 + krow] = f2bf(va.y);
      LA[(c4 + 2) * 32 + krow] = f2bf(va.z);
      LA[(c4 + 3) * 32 + krow] = f2bf(va.w);
      LB[(c4 + 0) * 32 + krow] = f2bf(vb.x);
      LB[(c4 + 1) * 32 + krow] = f2bf(vb.y);
      LB[(c4 + 2) * 32 + krow] = f2bf(vb.z);
      LB[(c4 + 3) * 32 + krow] = f2bf(vb.w);
    }
    __syncthreads();

    v16bf afr = ldfrag(&LA[(tr * 16 + lrow) * 32], khalf);
    v16bf b0  = ldfrag(&LB[((tcB + 0) * 16 + lrow) * 32], khalf);
    v16bf b1  = ldfrag(&LB[((tcB + 1) * 16 + lrow) * 32], khalf);
    acc[0] = __builtin_amdgcn_wmma_f32_16x16x32_bf16(false, afr, false, b0,
                                                     (short)0, acc[0], false, false);
    acc[1] = __builtin_amdgcn_wmma_f32_16x16x32_bf16(false, afr, false, b1,
                                                     (short)0, acc[1], false, false);
  }

  float* Gz = Gpart + (size_t)blockIdx.z * (DPROJ * DPROJ);
#pragma unroll
  for (int t = 0; t < 2; ++t)
#pragma unroll
    for (int r = 0; r < 8; ++r) {
      int p = ti * 64 + tr * 16 + r + 8 * khalf;
      int q = tj * 64 + (tcB + t) * 16 + lrow;
      Gz[p * DPROJ + q] = acc[t][r];   // deterministic: no atomics
    }
}

// ---------------------------------------------------------------------------
// 4) reduce split-K partials + covariance fixup + diagonal regularization
//    grid = 256, block = 256 (one thread per (i,j))
// ---------------------------------------------------------------------------
__global__ __launch_bounds__(256) void fixup(
    const float* __restrict__ Pxx, const float* __restrict__ Pxy,
    const float* __restrict__ Pyy, const float* __restrict__ sx,
    const float* __restrict__ sy, float* __restrict__ Axx,
    float* __restrict__ Cxy, float* __restrict__ Ayy) {
  int idx = blockIdx.x * 256 + threadIdx.x;     // 65536 elements
  int i = idx >> 8, j = idx & 255;
  float gxx = 0.f, gxy = 0.f, gyy = 0.f;
#pragma unroll
  for (int z = 0; z < NKCHUNK; ++z) {
    gxx += Pxx[z * (DPROJ * DPROJ) + idx];
    gxy += Pxy[z * (DPROJ * DPROJ) + idx];
    gyy += Pyy[z * (DPROJ * DPROJ) + idx];
  }
  const float invN  = 1.0f / (float)NSAMP;
  const float invN1 = 1.0f / ((float)NSAMP - 1.0f);
  float cxx = (gxx - sx[i] * sx[j] * invN) * invN1;
  float cxy = (gxy - sx[i] * sy[j] * invN) * invN1;
  float cyy = (gyy - sy[i] * sy[j] * invN) * invN1;
  Axx[idx] = cxx + (i == j ? EPSR : 0.f);
  Ayy[idx] = cyy + (i == j ? 2.0f * EPSR : 0.f);
  Cxy[idx] = cxy;
}

// ---------------------------------------------------------------------------
// 5) in-place Cholesky of a 256x256 SPD matrix (single block, 256 threads)
// ---------------------------------------------------------------------------
__global__ __launch_bounds__(256) void cholesky256(float* __restrict__ M) {
  __shared__ float red[256];
  __shared__ float diag;
  const int tid = threadIdx.x;
  for (int j = 0; j < 256; ++j) {
    float p = 0.f;
    for (int k = tid; k < j; k += 256) { float v = M[j * 256 + k]; p += v * v; }
    red[tid] = p; __syncthreads();
    for (int o = 128; o > 0; o >>= 1) {
      if (tid < o) red[tid] += red[tid + o];
      __syncthreads();
    }
    if (tid == 0) {
      float d = M[j * 256 + j] - red[0];
      diag = sqrtf(fmaxf(d, 1e-12f));
      M[j * 256 + j] = diag;
    }
    __syncthreads();
    for (int i = j + 1 + tid; i < 256; i += 256) {
      float s = M[i * 256 + j];
      for (int k = 0; k < j; ++k) s -= M[i * 256 + k] * M[j * 256 + k];
      M[i * 256 + j] = s / diag;
    }
    __syncthreads();
  }
}

// ---------------------------------------------------------------------------
// 6) SPD solve via L L^T: Out = (L L^T)^{-1} B (or B^T). One thread per RHS
//    column; the column recurrence is same-thread so no syncs needed.
// ---------------------------------------------------------------------------
__global__ __launch_bounds__(256) void chol_solve(
    const float* __restrict__ L, const float* __restrict__ Bsrc, int transB,
    float* __restrict__ Out) {
  const int c = threadIdx.x;
  for (int i = 0; i < 256; ++i) {               // forward: L y = b
    float s = transB ? Bsrc[c * 256 + i] : Bsrc[i * 256 + c];
    for (int k = 0; k < i; ++k) s -= L[i * 256 + k] * Out[k * 256 + c];
    Out[i * 256 + c] = s / L[i * 256 + i];
  }
  for (int i = 255; i >= 0; --i) {              // backward: L^T x = y
    float s = Out[i * 256 + c];
    for (int k = i + 1; k < 256; ++k) s -= L[k * 256 + i] * Out[k * 256 + c];
    Out[i * 256 + c] = s / L[i * 256 + i];
  }
}

// ---------------------------------------------------------------------------
// 7) loss = -trace(A1 @ A2)/D + lam*mean(Phi(mu/(sigma*sqrt2)))  (x and y)
// ---------------------------------------------------------------------------
__global__ __launch_bounds__(256) void finalize(
    const float* __restrict__ A1, const float* __restrict__ A2,
    const float* __restrict__ mux, const float* __restrict__ muy,
    float* __restrict__ out) {
  __shared__ float rT[256], rX[256], rY[256];
  const int tid = threadIdx.x;
  float t = 0.f;
  for (int idx = tid; idx < 256 * 256; idx += 256) {
    int i = idx >> 8, j = idx & 255;
    t += A1[i * 256 + j] * A2[j * 256 + i];
  }
  float rx = 0.f, ry = 0.f;
  const float scl = 1.41421356237f;   // 1/(SIGMA*sqrt(2)) with SIGMA=0.5
  for (int k = tid; k < DIN; k += 256) {
    rx += 0.5f * (1.0f + erff(mux[k] * scl));
    ry += 0.5f * (1.0f + erff(muy[k] * scl));
  }
  rT[tid] = t; rX[tid] = rx; rY[tid] = ry; __syncthreads();
  for (int o = 128; o > 0; o >>= 1) {
    if (tid < o) { rT[tid] += rT[tid + o]; rX[tid] += rX[tid + o]; rY[tid] += rY[tid + o]; }
    __syncthreads();
  }
  if (tid == 0)
    out[0] = -(rT[0] / (float)DPROJ)
             + 0.1f * (rX[0] / (float)DIN)
             + 0.1f * (rY[0] / (float)DIN);
}

// ---------------------------------------------------------------------------
extern "C" void kernel_launch(void* const* d_in, const int* in_sizes, int n_in,
                              void* d_out, int out_size, void* d_ws, size_t ws_size,
                              hipStream_t stream) {
  (void)in_sizes; (void)n_in; (void)out_size; (void)ws_size;
  const float* X   = (const float*)d_in[0];
  const float* Y   = (const float*)d_in[1];
  const float* mux = (const float*)d_in[2];
  const float* muy = (const float*)d_in[3];
  const float* Wx  = (const float*)d_in[4];
  const float* bx  = (const float*)d_in[5];
  const float* Wy  = (const float*)d_in[6];
  const float* by  = (const float*)d_in[7];

  // workspace layout (floats)
  float* Xh  = (float*)d_ws;                       // 65536*256
  float* Yh  = Xh  + (size_t)NSAMP * DPROJ;        // 65536*256
  float* sx  = Yh  + (size_t)NSAMP * DPROJ;        // 256
  float* sy  = sx  + DPROJ;                        // 256
  float* Pxx = sy  + DPROJ;                        // NKCHUNK*65536
  float* Pxy = Pxx + (size_t)NKCHUNK * DPROJ * DPROJ;
  float* Pyy = Pxy + (size_t)NKCHUNK * DPROJ * DPROJ;
  float* Axx = Pyy + (size_t)NKCHUNK * DPROJ * DPROJ;  // 65536 each below
  float* Ayy = Axx + DPROJ * DPROJ;
  float* Cxy = Ayy + DPROJ * DPROJ;
  float* A1  = Cxy + DPROJ * DPROJ;
  float* A2  = A1  + DPROJ * DPROJ;

  dim3 gProj(NSAMP / 128, DPROJ / 128);
  proj_gemm<<<gProj, 256, 0, stream>>>(X, mux, Wx, bx, Xh, DIN);
  proj_gemm<<<gProj, 256, 0, stream>>>(Y, muy, Wy, by, Yh, DIN);

  colsum<<<DPROJ, 256, 0, stream>>>(Xh, sx, NSAMP);
  colsum<<<DPROJ, 256, 0, stream>>>(Yh, sy, NSAMP);

  dim3 gGram(4, 4, NKCHUNK);
  gram_kernel<<<gGram, 256, 0, stream>>>(Xh, Xh, Pxx);
  gram_kernel<<<gGram, 256, 0, stream>>>(Xh, Yh, Pxy);
  gram_kernel<<<gGram, 256, 0, stream>>>(Yh, Yh, Pyy);

  fixup<<<DPROJ, 256, 0, stream>>>(Pxx, Pxy, Pyy, sx, sy, Axx, Cxy, Ayy);

  cholesky256<<<1, 256, 0, stream>>>(Axx);
  cholesky256<<<1, 256, 0, stream>>>(Ayy);

  // A1 = (Cyy+2e)^{-1} Cyx  (Cyx = Cxy^T -> transB=1)
  chol_solve<<<1, 256, 0, stream>>>(Ayy, Cxy, 1, A1);
  // A2 = (Cxx+e)^{-1} Cxy
  chol_solve<<<1, 256, 0, stream>>>(Axx, Cxy, 0, A2);

  finalize<<<1, 256, 0, stream>>>(A1, A2, mux, muy, (float*)d_out);
}
